// PCEncoderPose_34909494182011
// MI455X (gfx1250) — matrix operand order repaired and checked by reading, not compiled, not run on hardware
//
#include <hip/hip_runtime.h>

// ---------------------------------------------------------------------------
// PointNet++ encoder for MI455X (gfx1250, wave32, WMMA).
// All shared-MLP layers run on v_wmma_f32_16x16x32_f16 with f32 accumulate.
// ---------------------------------------------------------------------------

typedef __attribute__((ext_vector_type(16))) _Float16 v16h;
typedef __attribute__((ext_vector_type(8)))  _Float16 v8h;
typedef __attribute__((ext_vector_type(8)))  float    v8f;

#define NTHREADS 256
#define NWAVES   8
#define BATCH    16

// ---- WMMA fragment loaders -------------------------------------------------
// A operand (16x32 f16, M x K): row-major matrix, row stride ldw halfs.
// lanes 0-15 hold row r, K {0..7, 16..23}; lanes 16-31 row r, K {8..15, 24..31}
__device__ __forceinline__ v16h load_a_frag(const _Float16* W, int ldw, int r, int h) {
  const _Float16* p = W + r * ldw + h * 8;
  v8h lo = *(const v8h*)p;
  v8h hi = *(const v8h*)(p + 16);
  v16h out;
#pragma unroll
  for (int i = 0; i < 8; ++i) { out[i] = lo[i]; out[i + 8] = hi[i]; }
  return out;
}

// B operand (32x16 f16, K x N): activations stored column-major (col stride ldx).
// lanes 0-15 hold col r, K {0..7, 16..23}; lanes 16-31 col r, K {8..15, 24..31}
__device__ __forceinline__ v16h load_b_frag(const _Float16* X, int ldx, int r, int h) {
  const _Float16* p = X + r * ldx + h * 8;
  v8h lo = *(const v8h*)p;
  v8h hi = *(const v8h*)(p + 16);
  v16h out;
#pragma unroll
  for (int i = 0; i < 8; ++i) { out[i] = lo[i]; out[i + 8] = hi[i]; }
  return out;
}

// ---- One shared-MLP layer: Y = relu(diag(gs) * (W @ X) + be) --------------
// W: MxK row-major f16 (global, K padded to mult of 32)
// X: K x N activations, column-major f16 in LDS (col stride ldx >= K)
// If !MAXPOOL: writes Y column-major f16 into Xout (col stride ldo).
// If  MAXPOOL: max-reduces Y over columns into smax[] as uint bits (vals >= 0).
// The job loop bound is broadcast through readfirstlane so the branch is a
// scalar (SCC) branch: EXEC is guaranteed all-1s at every WMMA.
template <int M, int N, int K, bool MAXPOOL>
__device__ __forceinline__ void mlp_layer(const _Float16* __restrict__ W,
                                          const _Float16* X, int ldx,
                                          const float* __restrict__ gs,
                                          const float* __restrict__ be,
                                          _Float16* Xout, int ldo,
                                          unsigned* smax) {
  constexpr int MT = M / 16, NT = N / 16, KT = K / 32;
  const int wave = __builtin_amdgcn_readfirstlane(threadIdx.x >> 5);
  const int lane = threadIdx.x & 31;
  const int r = lane & 15;
  const int h = lane >> 4;
  for (int job = wave; job < MT * NT; job += NWAVES) {
    const int mt = job / NT, nt = job % NT;
    const _Float16* Wt = W + (mt * 16) * K;
    const _Float16* Xt = X + (nt * 16) * ldx;
    v8f acc = {};
    for (int kt = 0; kt < KT; ++kt) {
      v16h a  = load_a_frag(Wt + kt * 32, K,   r, h);
      v16h bm = load_b_frag(Xt + kt * 32, ldx, r, h);
      acc = __builtin_amdgcn_wmma_f32_16x16x32_f16(false, a, false, bm,
                                                   (short)0, acc, false, false);
    }
    // D layout: row = mt*16 + v + 8*h, col = nt*16 + (lane&15)
    const int mbase = mt * 16 + 8 * h;
    if constexpr (!MAXPOOL) {
      const int col = nt * 16 + r;
      v8h pack;
#pragma unroll
      for (int v = 0; v < 8; ++v) {
        float val = acc[v] * gs[mbase + v] + be[mbase + v];
        pack[v] = (_Float16)(val > 0.f ? val : 0.f);
      }
      *(v8h*)(Xout + col * ldo + mbase) = pack;
    } else {
#pragma unroll
      for (int v = 0; v < 8; ++v) {
        float val = acc[v] * gs[mbase + v] + be[mbase + v];
        val = val > 0.f ? val : 0.f;
#pragma unroll
        for (int m = 1; m < 16; m <<= 1)       // max over the tile's 16 cols
          val = fmaxf(val, __shfl_xor(val, m, 32));
        if (r == 0) atomicMax(&smax[mbase + v], __float_as_uint(val));
      }
    }
  }
  __syncthreads();
}

// Prefetch a weight panel toward the WGP (lowers to global_prefetch_b8).
__device__ __forceinline__ void prefetch_panel(const _Float16* p, int bytes) {
  for (int o = (int)threadIdx.x * 128; o < bytes; o += NTHREADS * 128)
    __builtin_prefetch((const char*)p + o, 0, 3);
}

// ---- Prep kernels ----------------------------------------------------------
__global__ void k_convert_w(const float* __restrict__ w, _Float16* __restrict__ out,
                            int cout, int cin, int kpad) {
  int t = blockIdx.x * blockDim.x + threadIdx.x;
  if (t >= cout * kpad) return;
  int o = t / kpad, k = t % kpad;
  out[t] = (k < cin) ? (_Float16)w[o * cin + k] : (_Float16)0.f;
}

__global__ void k_scale(const float* __restrict__ ga, float* __restrict__ gs, int n) {
  int t = blockIdx.x * blockDim.x + threadIdx.x;
  if (t < n) gs[t] = (float)((double)ga[t] * 0.99999500003749968);  // 1/sqrt(1+1e-5)
}

__global__ void k_zero(unsigned* p, int n) {
  int t = blockIdx.x * blockDim.x + threadIdx.x;
  if (t < n) p[t] = 0u;
}

// ---- Farthest point sampling: one workgroup per batch ----------------------
__global__ void __launch_bounds__(NTHREADS)
fps_kernel(const float* __restrict__ pts, int n, int stride,
           int npoint, int* __restrict__ fidx) {
  __shared__ float dist[4096];
  __shared__ float rv[NTHREADS];
  __shared__ int   ri[NTHREADS];
  const int b = blockIdx.x;
  const float* base = pts + (size_t)b * n * stride;
  for (int i = threadIdx.x; i < n; i += NTHREADS) dist[i] = 1e10f;
  int last = 0;
  __syncthreads();
  for (int it = 0; it < npoint; ++it) {
    if (threadIdx.x == 0) fidx[b * npoint + it] = last;
    const float lx = base[last * stride + 0];
    const float ly = base[last * stride + 1];
    const float lz = base[last * stride + 2];
    float bv = -1.f; int bi = 0;
    for (int i = threadIdx.x; i < n; i += NTHREADS) {
      float dx = base[i * stride + 0] - lx;
      float dy = base[i * stride + 1] - ly;
      float dz = base[i * stride + 2] - lz;
      float m = fminf(dist[i], dx * dx + dy * dy + dz * dz);
      dist[i] = m;
      if (m > bv) { bv = m; bi = i; }
    }
    rv[threadIdx.x] = bv; ri[threadIdx.x] = bi;
    __syncthreads();
    for (int s = NTHREADS / 2; s > 0; s >>= 1) {
      if (threadIdx.x < s) {
        float ov = rv[threadIdx.x + s]; int oi = ri[threadIdx.x + s];
        if (ov > rv[threadIdx.x] || (ov == rv[threadIdx.x] && oi < ri[threadIdx.x])) {
          rv[threadIdx.x] = ov; ri[threadIdx.x] = oi;
        }
      }
      __syncthreads();
    }
    last = ri[0];
    __syncthreads();
  }
}

// ---- Ball query: one thread per (batch, center) ----------------------------
__global__ void bq_kernel(const float* __restrict__ pts, int n, int stride, int npoint,
                          const int* __restrict__ fidx, float radius,
                          float* __restrict__ newxyz, int* __restrict__ gidx) {
  int t = blockIdx.x * blockDim.x + threadIdx.x;
  if (t >= BATCH * npoint) return;
  const int b = t / npoint;
  const float* base = pts + (size_t)b * n * stride;
  const int ci = fidx[t];
  const float cx = base[ci * stride + 0];
  const float cy = base[ci * stride + 1];
  const float cz = base[ci * stride + 2];
  newxyz[t * 3 + 0] = cx; newxyz[t * 3 + 1] = cy; newxyz[t * 3 + 2] = cz;
  int* g = gidx + (size_t)t * 64;
  const float r2 = radius * radius;
  int cnt = 0;
  for (int i = 0; i < n && cnt < 64; ++i) {
    float dx = base[i * stride + 0] - cx;
    float dy = base[i * stride + 1] - cy;
    float dz = base[i * stride + 2] - cz;
    if (dx * dx + dy * dy + dz * dz < r2) g[cnt++] = i;
  }
  const int pad = (cnt == 0) ? (n - 1) : g[0];
  for (int j = cnt; j < 64; ++j) g[j] = pad;
}

// ---- SA1: one WG per (batch, center); MLP 5(->32 pad)->64->64->128 ---------
__global__ void __launch_bounds__(NTHREADS)
sa1_kernel(const float* __restrict__ pc, const float* __restrict__ newxyz,
           const int* __restrict__ gidx,
           const _Float16* W1, const _Float16* W2, const _Float16* W3,
           const float* gs1, const float* be1,
           const float* gs2, const float* be2,
           const float* gs3, const float* be3,
           float* __restrict__ feat_out) {
  __shared__ __align__(16) _Float16 X0[64 * 32];
  __shared__ __align__(16) _Float16 X1[64 * 64];
  __shared__ __align__(16) _Float16 X2[64 * 64];
  __shared__ unsigned smax[128];
  const int b = blockIdx.x >> 9;
  const int s = blockIdx.x & 511;
  const int g = b * 512 + s;
  prefetch_panel(W1, 64 * 32 * 2);
  prefetch_panel(W2, 64 * 64 * 2);
  prefetch_panel(W3, 128 * 64 * 2);
  const float cx = newxyz[g * 3 + 0], cy = newxyz[g * 3 + 1], cz = newxyz[g * 3 + 2];
  for (int t = threadIdx.x; t < 64 * 32; t += NTHREADS) {
    const int j = t >> 5, c = t & 31;
    const int pi = gidx[g * 64 + j];
    const float* p = pc + (size_t)(b * 4096 + pi) * 5;
    float v;
    if (c == 0)      v = p[0] - cx;
    else if (c == 1) v = p[1] - cy;
    else if (c == 2) v = p[2] - cz;
    else if (c < 5)  v = p[c];           // feature channels
    else             v = 0.f;            // K padding
    X0[j * 32 + c] = (_Float16)v;
  }
  for (int t = threadIdx.x; t < 128; t += NTHREADS) smax[t] = 0u;
  __syncthreads();
  mlp_layer<64, 64, 32, false>(W1, X0, 32, gs1, be1, X1, 64, nullptr);
  mlp_layer<64, 64, 64, false>(W2, X1, 64, gs2, be2, X2, 64, nullptr);
  mlp_layer<128, 64, 64, true>(W3, X2, 64, gs3, be3, nullptr, 0, smax);
  for (int t = threadIdx.x; t < 128; t += NTHREADS)
    feat_out[(size_t)g * 128 + t] = __uint_as_float(smax[t]);
}

// ---- SA2: one WG per (batch, center); MLP 131(->160)->128->128->256 --------
__global__ void __launch_bounds__(NTHREADS)
sa2_kernel(const float* __restrict__ xyz1, const float* __restrict__ feat1,
           const float* __restrict__ newxyz, const int* __restrict__ gidx,
           const _Float16* W1, const _Float16* W2, const _Float16* W3,
           const float* gs1, const float* be1,
           const float* gs2, const float* be2,
           const float* gs3, const float* be3,
           float* __restrict__ feat_out) {
  __shared__ __align__(16) _Float16 X0[64 * 160];
  __shared__ __align__(16) _Float16 X1[64 * 128];
  __shared__ __align__(16) _Float16 X2[64 * 128];
  __shared__ unsigned smax[256];
  const int b = blockIdx.x >> 8;
  const int s = blockIdx.x & 255;
  const int g = b * 256 + s;
  prefetch_panel(W1, 128 * 160 * 2);
  prefetch_panel(W2, 128 * 128 * 2);
  prefetch_panel(W3, 256 * 128 * 2);
  const float cx = newxyz[g * 3 + 0], cy = newxyz[g * 3 + 1], cz = newxyz[g * 3 + 2];
  for (int t = threadIdx.x; t < 64 * 160; t += NTHREADS) {
    const int j = t / 160, c = t % 160;
    const int pi = gidx[g * 64 + j];
    float v;
    if (c < 3)        v = xyz1[(size_t)(b * 512 + pi) * 3 + c] - (c == 0 ? cx : c == 1 ? cy : cz);
    else if (c < 131) v = feat1[(size_t)(b * 512 + pi) * 128 + (c - 3)];
    else              v = 0.f;
    X0[j * 160 + c] = (_Float16)v;
  }
  for (int t = threadIdx.x; t < 256; t += NTHREADS) smax[t] = 0u;
  __syncthreads();
  mlp_layer<128, 64, 160, false>(W1, X0, 160, gs1, be1, X1, 128, nullptr);
  mlp_layer<128, 64, 128, false>(W2, X1, 128, gs2, be2, X2, 128, nullptr);
  mlp_layer<256, 64, 128, true>(W3, X2, 128, gs3, be3, nullptr, 0, smax);
  for (int t = threadIdx.x; t < 256; t += NTHREADS)
    feat_out[(size_t)g * 256 + t] = __uint_as_float(smax[t]);
}

// ---- SA3 (group-all): WG per 16-column tile; 259(->288)->256->512->1024 ----
__global__ void __launch_bounds__(NTHREADS)
sa3_kernel(const float* __restrict__ xyz2, const float* __restrict__ feat2,
           const _Float16* W1, const _Float16* W2, const _Float16* W3,
           const float* gs1, const float* be1,
           const float* gs2, const float* be2,
           const float* gs3, const float* be3,
           unsigned* __restrict__ feat3_u) {
  __shared__ __align__(16) _Float16 X0[16 * 288];
  __shared__ __align__(16) _Float16 X1[16 * 256];
  __shared__ __align__(16) _Float16 X2[16 * 512];
  const int b = blockIdx.x >> 4;
  const int ct = blockIdx.x & 15;
  prefetch_panel(W1, 256 * 288 * 2);
  for (int t = threadIdx.x; t < 16 * 288; t += NTHREADS) {
    const int j = t / 288, c = t % 288;
    const int n = ct * 16 + j;
    float v;
    if (c < 3)        v = xyz2[(size_t)(b * 256 + n) * 3 + c];
    else if (c < 259) v = feat2[(size_t)(b * 256 + n) * 256 + (c - 3)];
    else              v = 0.f;
    X0[j * 288 + c] = (_Float16)v;
  }
  __syncthreads();
  mlp_layer<256, 16, 288, false>(W1, X0, 288, gs1, be1, X1, 256, nullptr);
  mlp_layer<512, 16, 256, false>(W2, X1, 256, gs2, be2, X2, 512, nullptr);
  mlp_layer<1024, 16, 512, true>(W3, X2, 512, gs3, be3, nullptr, 0, feat3_u + (size_t)b * 1024);
}

// ---- Final FC: out[b][o] = feat3[b] . fc_w[o] + fc_b[o] --------------------
__global__ void fc_kernel(const float* __restrict__ feat3, const float* __restrict__ fcw,
                          const float* __restrict__ fcb, float* __restrict__ out) {
  int t = blockIdx.x * blockDim.x + threadIdx.x;
  if (t >= BATCH * 256) return;
  const int b = t >> 8, o = t & 255;
  const float* f = feat3 + (size_t)b * 1024;
  const float* w = fcw + (size_t)o * 1024;
  float acc = fcb[o];
  for (int k = 0; k < 1024; ++k) acc += f[k] * w[k];
  out[t] = acc;
}

// ---------------------------------------------------------------------------
extern "C" void kernel_launch(void* const* d_in, const int* in_sizes, int n_in,
                              void* d_out, int out_size, void* d_ws, size_t ws_size,
                              hipStream_t stream) {
  (void)in_sizes; (void)n_in; (void)out_size; (void)ws_size;
  const float* pc  = (const float*)d_in[0];
  const float* w_in[9]  = {(const float*)d_in[1],  (const float*)d_in[2],  (const float*)d_in[3],
                           (const float*)d_in[10], (const float*)d_in[11], (const float*)d_in[12],
                           (const float*)d_in[19], (const float*)d_in[20], (const float*)d_in[21]};
  const float* ga_in[9] = {(const float*)d_in[4],  (const float*)d_in[5],  (const float*)d_in[6],
                           (const float*)d_in[13], (const float*)d_in[14], (const float*)d_in[15],
                           (const float*)d_in[22], (const float*)d_in[23], (const float*)d_in[24]};
  const float* be_in[9] = {(const float*)d_in[7],  (const float*)d_in[8],  (const float*)d_in[9],
                           (const float*)d_in[16], (const float*)d_in[17], (const float*)d_in[18],
                           (const float*)d_in[25], (const float*)d_in[26], (const float*)d_in[27]};
  const float* fcw = (const float*)d_in[28];
  const float* fcb = (const float*)d_in[29];
  float* out = (float*)d_out;

  static const int couts[9] = {64, 64, 128, 128, 128, 256, 256, 512, 1024};
  static const int cins[9]  = {5,  64, 64,  131, 128, 128, 259, 256, 512};
  static const int kpads[9] = {32, 64, 64,  160, 128, 128, 288, 256, 512};

  char* ws = (char*)d_ws;
  size_t off = 0;
  auto alloc = [&](size_t bytes) -> void* {
    off = (off + 255) & ~(size_t)255;
    void* p = ws + off;
    off += bytes;
    return p;
  };
  int*   fidx1   = (int*)  alloc((size_t)BATCH * 512 * sizeof(int));
  float* newxyz1 = (float*)alloc((size_t)BATCH * 512 * 3 * sizeof(float));
  int*   gidx1   = (int*)  alloc((size_t)BATCH * 512 * 64 * sizeof(int));
  float* feat1   = (float*)alloc((size_t)BATCH * 512 * 128 * sizeof(float));
  int*   fidx2   = (int*)  alloc((size_t)BATCH * 256 * sizeof(int));
  float* newxyz2 = (float*)alloc((size_t)BATCH * 256 * 3 * sizeof(float));
  int*   gidx2   = (int*)  alloc((size_t)BATCH * 256 * 64 * sizeof(int));
  float* feat2   = (float*)alloc((size_t)BATCH * 256 * 256 * sizeof(float));
  float* feat3   = (float*)alloc((size_t)BATCH * 1024 * sizeof(float));
  _Float16* Wc[9];
  float* gsc[9];
  for (int i = 0; i < 9; ++i) {
    Wc[i]  = (_Float16*)alloc((size_t)couts[i] * kpads[i] * sizeof(_Float16));
    gsc[i] = (float*)   alloc((size_t)couts[i] * sizeof(float));
  }

  // Weight / BN-scale preparation.
  for (int i = 0; i < 9; ++i) {
    int n = couts[i] * kpads[i];
    k_convert_w<<<(n + 255) / 256, 256, 0, stream>>>(w_in[i], Wc[i], couts[i], cins[i], kpads[i]);
    k_scale<<<(couts[i] + 255) / 256, 256, 0, stream>>>(ga_in[i], gsc[i], couts[i]);
  }

  // Stage 1: FPS on raw points (stride 5), ball query r=0.2, fused WMMA MLP.
  fps_kernel<<<BATCH, NTHREADS, 0, stream>>>(pc, 4096, 5, 512, fidx1);
  bq_kernel<<<(BATCH * 512 + 127) / 128, 128, 0, stream>>>(pc, 4096, 5, 512, fidx1, 0.2f,
                                                           newxyz1, gidx1);
  sa1_kernel<<<BATCH * 512, NTHREADS, 0, stream>>>(pc, newxyz1, gidx1,
                                                   Wc[0], Wc[1], Wc[2],
                                                   gsc[0], be_in[0], gsc[1], be_in[1],
                                                   gsc[2], be_in[2], feat1);

  // Stage 2: FPS on 512 centers (stride 3), ball query r=0.4, fused WMMA MLP.
  fps_kernel<<<BATCH, NTHREADS, 0, stream>>>(newxyz1, 512, 3, 256, fidx2);
  bq_kernel<<<(BATCH * 256 + 127) / 128, 128, 0, stream>>>(newxyz1, 512, 3, 256, fidx2, 0.4f,
                                                           newxyz2, gidx2);
  sa2_kernel<<<BATCH * 256, NTHREADS, 0, stream>>>(newxyz1, feat1, newxyz2, gidx2,
                                                   Wc[3], Wc[4], Wc[5],
                                                   gsc[3], be_in[3], gsc[4], be_in[4],
                                                   gsc[5], be_in[5], feat2);

  // Stage 3: group-all fused WMMA MLP with global maxpool (ReLU -> uint max).
  k_zero<<<(BATCH * 1024 + 255) / 256, 256, 0, stream>>>((unsigned*)feat3, BATCH * 1024);
  sa3_kernel<<<BATCH * 16, NTHREADS, 0, stream>>>(newxyz2, feat2,
                                                  Wc[6], Wc[7], Wc[8],
                                                  gsc[6], be_in[6], gsc[7], be_in[7],
                                                  gsc[8], be_in[8], (unsigned*)feat3);

  // Final FC (tiny: 16x256x1024 f32).
  fc_kernel<<<(BATCH * 256 + 255) / 256, 256, 0, stream>>>(feat3, fcw, fcb, out);
}
